// TestAllReduceRMSNormModel_45354854645975
// MI455X (gfx1250) — matrix operand order repaired and checked by reading, not compiled, not run on hardware
//
#include <hip/hip_runtime.h>

typedef __bf16 bf16_t;
typedef __attribute__((ext_vector_type(16))) __bf16 v16bf;
typedef __attribute__((ext_vector_type(8)))  __bf16 v8bf;
typedef __attribute__((ext_vector_type(4)))  __bf16 v4bf;
typedef __attribute__((ext_vector_type(8)))  float  v8f;
typedef __attribute__((ext_vector_type(4)))  int    v4i;

#define BLK_M 128
#define BLK_N 256
#define TILE_K 32
#define LDSS   40   // padded LDS row stride in bf16 (80B) to spread banks

#define AS1 __attribute__((address_space(1)))
#define AS3 __attribute__((address_space(3)))

#if defined(__gfx1250__) && __has_builtin(__builtin_amdgcn_global_load_async_to_lds_b128)
#define USE_ASYNC 1
#else
#define USE_ASYNC 0
#endif

#if USE_ASYNC
__device__ __forceinline__ void async_b128(const bf16_t* g, bf16_t* l) {
  // GLOBAL_LOAD_ASYNC_TO_LDS_B128: per-lane global addr -> per-lane LDS addr,
  // tracked by ASYNCcnt (no VGPR data round-trip).
  __builtin_amdgcn_global_load_async_to_lds_b128((AS1 v4i*)g, (AS3 v4i*)l, 0, 0);
}
__device__ __forceinline__ void wait_async() {
#if __has_builtin(__builtin_amdgcn_s_wait_asynccnt)
  __builtin_amdgcn_s_wait_asynccnt(0);
#else
  asm volatile("s_wait_asynccnt 0x0" ::: "memory");
#endif
}
#endif

// ---------------------------------------------------------------------------
// Kernel 1: z = relu(x); resid = z; y = bf16(rmsnorm(z, g))
// ---------------------------------------------------------------------------
__global__ __launch_bounds__(256)
void relu_rms_kernel(const float* __restrict__ x, const float* __restrict__ g,
                     float* __restrict__ resid, bf16_t* __restrict__ y, int H) {
  const int row = blockIdx.x;
  const int t   = threadIdx.x;
  const float4* x4 = (const float4*)(x + (size_t)row * H);
  float4*       r4 = (float4*)(resid + (size_t)row * H);
  float4 v[4];
  float acc = 0.f;
#pragma unroll
  for (int j = 0; j < 4; ++j) {
    float4 a = x4[t + j * 256];
    a.x = fmaxf(a.x, 0.f); a.y = fmaxf(a.y, 0.f);
    a.z = fmaxf(a.z, 0.f); a.w = fmaxf(a.w, 0.f);
    v[j] = a;
    r4[t + j * 256] = a;
    acc += a.x * a.x + a.y * a.y + a.z * a.z + a.w * a.w;
  }
  __shared__ float red[256];
  red[t] = acc;
  __syncthreads();
#pragma unroll
  for (int s = 128; s > 0; s >>= 1) {
    if (t < s) red[t] += red[t + s];
    __syncthreads();
  }
  const float rs = rsqrtf(red[0] / (float)H + 1e-6f);
  const float4* g4 = (const float4*)g;
  v4bf* y4 = (v4bf*)(y + (size_t)row * H);
#pragma unroll
  for (int j = 0; j < 4; ++j) {
    float4 gv = g4[t + j * 256];
    v4bf o;
    o[0] = (bf16_t)(v[j].x * rs * gv.x);
    o[1] = (bf16_t)(v[j].y * rs * gv.y);
    o[2] = (bf16_t)(v[j].z * rs * gv.z);
    o[3] = (bf16_t)(v[j].w * rs * gv.w);
    y4[t + j * 256] = o;
  }
}

// ---------------------------------------------------------------------------
// Kernel 2: fused residual add + rmsnorm (bf16 out for next GEMM, or f32 final)
// ---------------------------------------------------------------------------
__global__ __launch_bounds__(256)
void fused_add_rms_kernel(const float* __restrict__ xin, float* __restrict__ resid,
                          const float* __restrict__ g, bf16_t* __restrict__ ybf,
                          float* __restrict__ yf, int H) {
  const int row = blockIdx.x;
  const int t   = threadIdx.x;
  const float4* x4 = (const float4*)(xin + (size_t)row * H);
  float4*       r4 = (float4*)(resid + (size_t)row * H);
  float4 v[4];
  float acc = 0.f;
#pragma unroll
  for (int j = 0; j < 4; ++j) {
    float4 a = x4[t + j * 256];
    float4 r = r4[t + j * 256];
    a.x += r.x; a.y += r.y; a.z += r.z; a.w += r.w;
    v[j] = a;
    if (ybf) r4[t + j * 256] = a;   // residual only needed by later layers
    acc += a.x * a.x + a.y * a.y + a.z * a.z + a.w * a.w;
  }
  __shared__ float red[256];
  red[t] = acc;
  __syncthreads();
#pragma unroll
  for (int s = 128; s > 0; s >>= 1) {
    if (t < s) red[t] += red[t + s];
    __syncthreads();
  }
  const float rs = rsqrtf(red[0] / (float)H + 1e-6f);
  const float4* g4 = (const float4*)g;
  if (ybf) {
    v4bf* y4 = (v4bf*)(ybf + (size_t)row * H);
#pragma unroll
    for (int j = 0; j < 4; ++j) {
      float4 gv = g4[t + j * 256];
      v4bf o;
      o[0] = (bf16_t)(v[j].x * rs * gv.x);
      o[1] = (bf16_t)(v[j].y * rs * gv.y);
      o[2] = (bf16_t)(v[j].z * rs * gv.z);
      o[3] = (bf16_t)(v[j].w * rs * gv.w);
      y4[t + j * 256] = o;
    }
  } else {
    float4* y4 = (float4*)(yf + (size_t)row * H);
#pragma unroll
    for (int j = 0; j < 4; ++j) {
      float4 gv = g4[t + j * 256];
      float4 o;
      o.x = v[j].x * rs * gv.x; o.y = v[j].y * rs * gv.y;
      o.z = v[j].z * rs * gv.z; o.w = v[j].w * rs * gv.w;
      y4[t + j * 256] = o;
    }
  }
}

// ---------------------------------------------------------------------------
// Kernel 3: W (f32, KxN row-major) -> Wt (bf16, NxK) via 32x32 LDS tile
// ---------------------------------------------------------------------------
__global__ __launch_bounds__(256)
void wcvt_kernel(const float* __restrict__ W, bf16_t* __restrict__ Wt, int K, int N) {
  __shared__ float tile[32][33];
  const int tx = threadIdx.x & 31;
  const int ty = threadIdx.x >> 5;
  const int kb = blockIdx.y * 32;
  const int nb = blockIdx.x * 32;
#pragma unroll
  for (int j = 0; j < 4; ++j)
    tile[ty + j * 8][tx] = W[(size_t)(kb + ty + j * 8) * N + nb + tx];
  __syncthreads();
#pragma unroll
  for (int j = 0; j < 4; ++j)
    Wt[(size_t)(nb + ty + j * 8) * K + kb + tx] = (bf16_t)tile[tx][ty + j * 8];
}

// ---------------------------------------------------------------------------
// GEMM compute step: 4x4 WMMA tiles per wave from one LDS stage buffer.
// Fragment layouts per CDNA5 ISA 7.12.2 (A 16x32 bf16, B 32x16 bf16).
// ---------------------------------------------------------------------------
__device__ __forceinline__ void compute_tile(const bf16_t* __restrict__ sA,
                                             const bf16_t* __restrict__ sB,
                                             v8f (&acc)[4][4], int wrow, int wcol,
                                             int mr, int aHalf, int bHalf) {
  v16bf fa[4], fb2[4];
#pragma unroll
  for (int mt = 0; mt < 4; ++mt) {
    const bf16_t* p = sA + (wrow * 64 + mt * 16 + mr) * LDSS + aHalf;
    v8bf lo = *(const v8bf*)p;
    v8bf hi = *(const v8bf*)(p + 16);
    fa[mt] = __builtin_shufflevector(lo, hi, 0, 1, 2, 3, 4, 5, 6, 7,
                                              8, 9, 10, 11, 12, 13, 14, 15);
  }
#pragma unroll
  for (int nt = 0; nt < 4; ++nt) {
    const bf16_t* p = sB + (wcol * 64 + nt * 16 + mr) * LDSS + bHalf;
    v8bf lo = *(const v8bf*)p;
    v8bf hi = *(const v8bf*)(p + 16);
    fb2[nt] = __builtin_shufflevector(lo, hi, 0, 1, 2, 3, 4, 5, 6, 7,
                                               8, 9, 10, 11, 12, 13, 14, 15);
  }
#pragma unroll
  for (int mt = 0; mt < 4; ++mt)
#pragma unroll
    for (int nt = 0; nt < 4; ++nt)
      acc[mt][nt] = __builtin_amdgcn_wmma_f32_16x16x32_bf16(
          false, fa[mt], false, fb2[nt], (short)0, acc[mt][nt], false, false);
}

// ---------------------------------------------------------------------------
// Kernel 4: C[M,N] f32 = A[M,K] bf16 * Bt[N,K] bf16.
// Block tile 128x256, 8 wave32 waves (2x4), wave tile 64x64 (4x4 WMMA tiles).
// Async global->LDS double-buffered pipeline when available.
// ---------------------------------------------------------------------------
__global__ __launch_bounds__(256)
void gemm_bf16_kernel(const bf16_t* __restrict__ A, const bf16_t* __restrict__ Bt,
                      float* __restrict__ C, int M, int N, int K) {
  __shared__ bf16_t smA[2][BLK_M * LDSS];   // 2 x 10 KB
  __shared__ bf16_t smB[2][BLK_N * LDSS];   // 2 x 20 KB

  const int tid  = threadIdx.x;
  const int lane = tid & 31;
  const int wid  = tid >> 5;
  const int wrow = wid >> 2;   // 0..1 -> 64-row slab
  const int wcol = wid & 3;    // 0..3 -> 64-col slab
  const int mBlk = blockIdx.y * BLK_M;
  const int nBlk = blockIdx.x * BLK_N;

  // staging coords: A -> 2 threads/row x 16 bf16; B -> 1 thread/row x 32 bf16
  const int aRow = tid >> 1;
  const int aKh  = (tid & 1) * 16;
  const bf16_t* gA = A  + (size_t)(mBlk + aRow) * K + aKh;
  const bf16_t* gB = Bt + (size_t)(nBlk + tid) * K;
  bf16_t* sA0 = &smA[0][aRow * LDSS + aKh];
  bf16_t* sA1 = &smA[1][aRow * LDSS + aKh];
  bf16_t* sB0 = &smB[0][tid * LDSS];
  bf16_t* sB1 = &smB[1][tid * LDSS];

  v8f acc[4][4];
  const v8f vzero = {0.f, 0.f, 0.f, 0.f, 0.f, 0.f, 0.f, 0.f};
#pragma unroll
  for (int mt = 0; mt < 4; ++mt)
#pragma unroll
    for (int nt = 0; nt < 4; ++nt) acc[mt][nt] = vzero;

  const int mr    = lane & 15;
  const int aHalf = (lane >> 4) * 8;    // A lane<16: K{0..7,16..23}; else {8..15,24..31}
  const int bHalf = (lane >> 4) * 8;    // B lane<16: K{0..7,16..23}; else {8..15,24..31}
  const int nk    = K / TILE_K;

#if USE_ASYNC
  // prime buffer 0
  async_b128(gA, sA0);
  async_b128(gA + 8, sA0 + 8);
#pragma unroll
  for (int j = 0; j < 4; ++j) async_b128(gB + j * 8, sB0 + j * 8);

  for (int kt = 0; kt < nk; ++kt) {
    wait_async();
    __syncthreads();
    const int cur = kt & 1;
    if (kt + 1 < nk) {
      const bf16_t* pA = gA + (size_t)(kt + 1) * TILE_K;
      const bf16_t* pB = gB + (size_t)(kt + 1) * TILE_K;
      bf16_t* dA = cur ? sA0 : sA1;
      bf16_t* dB = cur ? sB0 : sB1;
      async_b128(pA, dA);
      async_b128(pA + 8, dA + 8);
#pragma unroll
      for (int j = 0; j < 4; ++j) async_b128(pB + j * 8, dB + j * 8);
    }
    compute_tile(smA[cur], smB[cur], acc, wrow, wcol, mr, aHalf, bHalf);
  }
#else
  for (int kt = 0; kt < nk; ++kt) {
    const int k0 = kt * TILE_K;
    uint4 a0 = *(const uint4*)(gA + k0);
    uint4 a1 = *(const uint4*)(gA + k0 + 8);
    uint4 b0 = *(const uint4*)(gB + k0);
    uint4 b1 = *(const uint4*)(gB + k0 + 8);
    uint4 b2 = *(const uint4*)(gB + k0 + 16);
    uint4 b3 = *(const uint4*)(gB + k0 + 24);
    if (k0 + TILE_K < K) {
      __builtin_prefetch(gA + k0 + TILE_K, 0, 1);
      __builtin_prefetch(gB + k0 + TILE_K, 0, 1);
    }
    *(uint4*)sA0 = a0; *((uint4*)sA0 + 1) = a1;
    ((uint4*)sB0)[0] = b0; ((uint4*)sB0)[1] = b1;
    ((uint4*)sB0)[2] = b2; ((uint4*)sB0)[3] = b3;
    __syncthreads();
    compute_tile(smA[0], smB[0], acc, wrow, wcol, mr, aHalf, bHalf);
    __syncthreads();
  }
  (void)sA1; (void)sB1;
#endif

  // C layout: VGPR r -> row r + 8*(lane/16); col = lane%16
  const int rHalf = (lane >> 4) * 8;
#pragma unroll
  for (int mt = 0; mt < 4; ++mt) {
#pragma unroll
    for (int nt = 0; nt < 4; ++nt) {
      const int row0 = mBlk + wrow * 64 + mt * 16 + rHalf;
      const int col  = nBlk + wcol * 64 + nt * 16 + mr;
      float* cp = C + (size_t)row0 * N + col;
#pragma unroll
      for (int r = 0; r < 8; ++r) cp[(size_t)r * N] = acc[mt][nt][r];
    }
  }
}

// ---------------------------------------------------------------------------
extern "C" void kernel_launch(void* const* d_in, const int* in_sizes, int n_in,
                              void* d_out, int out_size, void* d_ws, size_t ws_size,
                              hipStream_t stream) {
  const float* x  = (const float*)d_in[0];
  const float* g0 = (const float*)d_in[1];
  const float* gs[3] = {(const float*)d_in[2], (const float*)d_in[3], (const float*)d_in[4]};
  const float* w[3]  = {(const float*)d_in[5], (const float*)d_in[6], (const float*)d_in[7]};

  const int H = in_sizes[1];        // 4096
  const int M = in_sizes[0] / H;    // 8192
  float* out = (float*)d_out;

  // workspace: resid f32 [M*H] | y bf16 [M*H] | Wt bf16 [H*H]
  float*  resid = (float*)d_ws;
  bf16_t* ybf   = (bf16_t*)((char*)d_ws + (size_t)M * H * 4);
  bf16_t* wt    = (bf16_t*)((char*)d_ws + (size_t)M * H * 4 + (size_t)M * H * 2);

  dim3 blk(256);
  dim3 rowGrid(M);
  dim3 tGrid(H / 32, H / 32);
  dim3 gGrid(H / BLK_N, M / BLK_M);

  relu_rms_kernel<<<rowGrid, blk, 0, stream>>>(x, g0, resid, ybf, H);

  for (int l = 0; l < 3; ++l) {
    wcvt_kernel<<<tGrid, blk, 0, stream>>>(w[l], wt, H, H);
    gemm_bf16_kernel<<<gGrid, blk, 0, stream>>>(ybf, wt, out, M, H, H);
    const bool last = (l == 2);
    fused_add_rms_kernel<<<rowGrid, blk, 0, stream>>>(
        out, resid, gs[l], last ? (bf16_t*)nullptr : ybf, last ? out : nullptr, H);
  }
}
